// Entropic_OT_Loss_57801669869626
// MI455X (gfx1250) — compile-verified, matrix operand order checked by reading, not run on hardware
//
#include <hip/hip_runtime.h>

typedef __attribute__((ext_vector_type(16))) __bf16 v16bf;
typedef __attribute__((ext_vector_type(8)))  float  v8f;

#define N_EMB 4096
#define D_EMB 512

// GEMM tiling
#define TM 128
#define TN 64
#define BK 32
#define NK (D_EMB / BK)
#define LSTR 48                       // LDS row stride in halfwords: 96B rows, 16B aligned
#define BUFHW (2*TM*LSTR + 2*TN*LSTR) // halfwords per double-buffer slot (36KB)
#define OFF_ALO (TM*LSTR)             // halfword offsets within a buffer slot
#define OFF_BHI (2*TM*LSTR)
#define OFF_BLO (2*TM*LSTR + TN*LSTR)

__device__ __forceinline__ unsigned short f2bf(float f) {
  unsigned int x = __float_as_uint(f);
  unsigned int r = (x + 0x7FFFu + ((x >> 16) & 1u)) >> 16;  // RNE
  return (unsigned short)r;
}
__device__ __forceinline__ float bf2f(unsigned short h) {
  return __uint_as_float(((unsigned int)h) << 16);
}

// one 16B global -> LDS async copy (per lane), tracked by ASYNCcnt
#define ASYNC_CP16(ldsaddr, gptr)                                              \
  asm volatile("global_load_async_to_lds_b128 %0, %1, off"                     \
               :: "v"(ldsaddr),                                                \
                  "v"((unsigned long long)(uintptr_t)(gptr))                   \
               : "memory")
#define WAIT_ASYNC0() asm volatile("s_wait_asynccnt 0x0" ::: "memory")

// ---------------- fp32 -> (hi,lo) bf16 split ----------------
__global__ void convert_hilo(const float* __restrict__ x,
                             unsigned short* __restrict__ hi,
                             unsigned short* __restrict__ lo, int n) {
  int i = blockIdx.x * blockDim.x + threadIdx.x;
  if (i < n) {
    float v = x[i];
    unsigned short h = f2bf(v);
    hi[i] = h;
    lo[i] = f2bf(v - bf2f(h));
  }
}

union FragU { uint4 q[2]; v16bf v; };

// ---------------- S = img @ txt^T via split-bf16 WMMA, async double-buffered LDS ----------------
__global__ __launch_bounds__(256)
void gemm_wmma(const unsigned short* __restrict__ Ahi,
               const unsigned short* __restrict__ Alo,
               const unsigned short* __restrict__ Bhi,
               const unsigned short* __restrict__ Blo,
               float* __restrict__ S) {
  __shared__ __align__(16) unsigned short smem[2 * BUFHW];   // 72KB (320KB/WGP available)

  const int tid   = threadIdx.x;
  const int wid   = tid >> 5;
  const int lane  = tid & 31;
  const int lrow  = lane & 15;
  const int khalf = (lane >> 4) << 3;   // 0 or 8 (K-half select per ISA A/B layout)

  const int waveM = wid >> 1;  // 0..3
  const int waveN = wid & 1;   // 0..1

  const int rowA0 = blockIdx.y * TM;   // img rows
  const int rowB0 = blockIdx.x * TN;   // txt rows

  // ---- per-thread copy assignment: 6 x 16B chunks per K-step ----
  const int rA1 = tid >> 2;            // 0..63
  const int rA2 = 64 + rA1;            // 64..127
  const int ch  = (tid & 3) << 3;      // halfword offset 0/8/16/24 within BK
  const int rB  = tid >> 2;            // 0..63

  const unsigned ldsBase = (unsigned)(uintptr_t)smem;
  const unsigned dA1 = ldsBase + (unsigned)(rA1 * LSTR + ch) * 2;
  const unsigned dA2 = ldsBase + (unsigned)(rA2 * LSTR + ch) * 2;
  const unsigned dB  = ldsBase + (unsigned)(rB  * LSTR + ch) * 2;

  const unsigned short* gAhi1 = Ahi + (size_t)(rowA0 + rA1) * D_EMB + ch;
  const unsigned short* gAhi2 = Ahi + (size_t)(rowA0 + rA2) * D_EMB + ch;
  const unsigned short* gAlo1 = Alo + (size_t)(rowA0 + rA1) * D_EMB + ch;
  const unsigned short* gAlo2 = Alo + (size_t)(rowA0 + rA2) * D_EMB + ch;
  const unsigned short* gBhi  = Bhi + (size_t)(rowB0 + rB ) * D_EMB + ch;
  const unsigned short* gBlo  = Blo + (size_t)(rowB0 + rB ) * D_EMB + ch;

  auto issue_tile = [&](int buf, int kk) {
    const unsigned b = (unsigned)buf * (BUFHW * 2u);
    ASYNC_CP16(dA1 + b,                   gAhi1 + kk);
    ASYNC_CP16(dA2 + b,                   gAhi2 + kk);
    ASYNC_CP16(dA1 + b + OFF_ALO * 2u,    gAlo1 + kk);
    ASYNC_CP16(dA2 + b + OFF_ALO * 2u,    gAlo2 + kk);
    ASYNC_CP16(dB  + b + OFF_BHI * 2u,    gBhi  + kk);
    ASYNC_CP16(dB  + b + OFF_BLO * 2u,    gBlo  + kk);
  };

  v8f acc00 = 0, acc01 = 0, acc10 = 0, acc11 = 0;

  // prologue: fill buffer 0
  issue_tile(0, 0);
  WAIT_ASYNC0();
  __syncthreads();

  const int ar0 = waveM * 32 + lrow, ar1 = ar0 + 16;
  const int br0 = waveN * 32 + lrow, br1 = br0 + 16;

  for (int ks = 0; ks < NK; ++ks) {
    const int cur = ks & 1;
    if (ks + 1 < NK) issue_tile(cur ^ 1, (ks + 1) * BK);  // overlap copy with WMMA

    const unsigned short* bufp = smem + cur * BUFHW;
    auto ldfrag = [&](const unsigned short* base, int row) -> v16bf {
      FragU f;
      const unsigned short* p = base + row * LSTR + khalf;
      f.q[0] = *(const uint4*)p;          // K = khalf .. khalf+7
      f.q[1] = *(const uint4*)(p + 16);   // K = khalf+16 .. khalf+23
      return f.v;
    };

    v16bf ah0 = ldfrag(bufp,           ar0), ah1 = ldfrag(bufp,           ar1);
    v16bf al0 = ldfrag(bufp + OFF_ALO, ar0), al1 = ldfrag(bufp + OFF_ALO, ar1);
    v16bf bh0 = ldfrag(bufp + OFF_BHI, br0), bh1 = ldfrag(bufp + OFF_BHI, br1);
    v16bf bl0 = ldfrag(bufp + OFF_BLO, br0), bl1 = ldfrag(bufp + OFF_BLO, br1);

#define WMMA3(ACC, AH, AL, BH, BL)                                                                    \
    ACC = __builtin_amdgcn_wmma_f32_16x16x32_bf16(false, AH, false, BH, (short)0, ACC, false, false); \
    ACC = __builtin_amdgcn_wmma_f32_16x16x32_bf16(false, AH, false, BL, (short)0, ACC, false, false); \
    ACC = __builtin_amdgcn_wmma_f32_16x16x32_bf16(false, AL, false, BH, (short)0, ACC, false, false);

    WMMA3(acc00, ah0, al0, bh0, bl0)
    WMMA3(acc01, ah0, al0, bh1, bl1)
    WMMA3(acc10, ah1, al1, bh0, bl0)
    WMMA3(acc11, ah1, al1, bh1, bl1)
#undef WMMA3

    if (ks + 1 < NK) {
      WAIT_ASYNC0();      // next tile landed in LDS (this wave's copies)
      __syncthreads();    // all waves' copies visible; prior reads of other buffer done
    }
  }

  // C/D layout: VGPR g -> M = g + (lane>=16 ? 8 : 0); N = lane % 16
  const int mh = (lane >> 4) << 3;
  const int m0 = rowA0 + waveM * 32;
  const int n0 = rowB0 + waveN * 32;
  for (int g = 0; g < 8; ++g) {
    S[(size_t)(m0 +  0 + mh + g) * N_EMB + (n0 +  0 + lrow)] = acc00[g];
    S[(size_t)(m0 +  0 + mh + g) * N_EMB + (n0 + 16 + lrow)] = acc01[g];
    S[(size_t)(m0 + 16 + mh + g) * N_EMB + (n0 +  0 + lrow)] = acc10[g];
    S[(size_t)(m0 + 16 + mh + g) * N_EMB + (n0 + 16 + lrow)] = acc11[g];
  }
}

// ---------------- Sinkhorn helpers (all scans of S are row-major/coalesced) ----------------
__global__ void zero_vec(float* p, int n) {
  int i = blockIdx.x * blockDim.x + threadIdx.x;
  if (i < n) p[i] = 0.f;
}
__global__ void init_uv(float* u, float* v, int n) {
  int i = blockIdx.x * blockDim.x + threadIdx.x;
  if (i < n) { u[i] = 1.f / (float)n; v[i] = 1.f / (float)n; }
}
__global__ void vdiv_b(float* v, const float* t, int n) {
  int i = blockIdx.x * blockDim.x + threadIdx.x;
  if (i < n) v[i] = (1.f / (float)n) / t[i];
}

// out[j] += sum_i exp(sign*100*(S[i,j]-1)) * w[i]   (column result, atomic partials)
#define CS_ROWS 128
__global__ __launch_bounds__(256)
void colsum_exp(const float* __restrict__ S, const float* __restrict__ w,
                float* __restrict__ out, float sign) {
  const int j  = blockIdx.x * 256 + threadIdx.x;
  const int i0 = blockIdx.y * CS_ROWS;
  float acc = 0.f;
  for (int i = i0; i < i0 + CS_ROWS; ++i)
    acc += __expf(sign * 100.f * (S[(size_t)i * N_EMB + j] - 1.f)) * w[i];
  atomicAdd(&out[j], acc);
}

// out[i] = sum_j exp(sign*100*(S[i,j]-1)) * w[j]   (row result, one wave per row)
__global__ __launch_bounds__(256)
void rowdot_exp(const float* __restrict__ S, const float* __restrict__ w,
                float* __restrict__ out, float sign) {
  const int lane = threadIdx.x & 31;
  const int row  = blockIdx.x * 8 + (threadIdx.x >> 5);
  const float* Sr = S + (size_t)row * N_EMB;
  float acc = 0.f;
  for (int j = lane; j < N_EMB; j += 32)
    acc += __expf(sign * 100.f * (Sr[j] - 1.f)) * w[j];
  for (int o = 16; o; o >>= 1) acc += __shfl_xor(acc, o, 32);
  if (lane == 0) out[row] = acc;
}

// ---------------- losses ----------------
// image loss: rows of S.  plan[i,j] = u[i]*v[i]*K[i,j];  nll_i = -(p_ii - lse_j p_ij)
__global__ __launch_bounds__(256)
void loss_rows(const float* __restrict__ S, const float* __restrict__ u,
               const float* __restrict__ v, float* __restrict__ out, float scale) {
  const int lane = threadIdx.x & 31;
  const int row  = blockIdx.x * 8 + (threadIdx.x >> 5);
  const float c = u[row] * v[row];
  const float* Sr = S + (size_t)row * N_EMB;
  float m = -INFINITY, s = 0.f, pd = 0.f;
  for (int j = lane; j < N_EMB; j += 32) {
    float p = c * __expf(100.f * (Sr[j] - 1.f));
    if (j == row) pd = p;
    if (p > m) { s = s * __expf(m - p) + 1.f; m = p; }
    else       { s += __expf(p - m); }
  }
  for (int o = 16; o; o >>= 1) {
    float m2 = __shfl_xor(m, o, 32), s2 = __shfl_xor(s, o, 32);
    float M = fmaxf(m, m2);
    s = s * __expf(m - M) + s2 * __expf(m2 - M);
    m = M;
    pd += __shfl_xor(pd, o, 32);
  }
  if (lane == 0) atomicAdd(out, -(pd - (m + __logf(s))) * scale);
}

// text loss: row i of S^T == column i of S; thread-per-column, coalesced across lanes
__global__ __launch_bounds__(256)
void loss_cols(const float* __restrict__ S, const float* __restrict__ u,
               const float* __restrict__ v, float* __restrict__ out, float scale) {
  const int i = blockIdx.x * 256 + threadIdx.x;
  const float c = u[i] * v[i];
  float m = -INFINITY, s = 0.f, pd = 0.f;
  for (int j = 0; j < N_EMB; ++j) {
    float p = c * __expf(100.f * (S[(size_t)j * N_EMB + i] - 1.f));
    if (j == i) pd = p;
    if (p > m) { s = s * __expf(m - p) + 1.f; m = p; }
    else       { s += __expf(p - m); }
  }
  atomicAdd(out, -(pd - (m + __logf(s))) * scale);
}

// ---------------- host-side orchestration ----------------
extern "C" void kernel_launch(void* const* d_in, const int* in_sizes, int n_in,
                              void* d_out, int out_size, void* d_ws, size_t ws_size,
                              hipStream_t stream) {
  (void)in_sizes; (void)n_in; (void)out_size; (void)ws_size;
  const float* img = (const float*)d_in[0];
  const float* txt = (const float*)d_in[1];
  float* out = (float*)d_out;

  char* ws = (char*)d_ws;
  float*          Smat = (float*)ws;                              // 64 MB
  unsigned short* Ahi  = (unsigned short*)(ws + ((size_t)64 << 20));
  unsigned short* Alo  = Ahi + (size_t)N_EMB * D_EMB;
  unsigned short* Bhi  = Alo + (size_t)N_EMB * D_EMB;
  unsigned short* Blo  = Bhi + (size_t)N_EMB * D_EMB;
  float*          tvec = (float*)(Blo + (size_t)N_EMB * D_EMB);
  float*          uvec = tvec + N_EMB;
  float*          vvec = uvec + N_EMB;

  const int nElem = N_EMB * D_EMB;
  const float scale = 1.f / (2.f * (float)N_EMB);

  zero_vec<<<1, 32, 0, stream>>>(out, 1);
  convert_hilo<<<(nElem + 255) / 256, 256, 0, stream>>>(img, Ahi, Alo, nElem);
  convert_hilo<<<(nElem + 255) / 256, 256, 0, stream>>>(txt, Bhi, Blo, nElem);

  gemm_wmma<<<dim3(N_EMB / TN, N_EMB / TM), 256, 0, stream>>>(Ahi, Alo, Bhi, Blo, Smat);

  const dim3 csGrid(N_EMB / 256, N_EMB / CS_ROWS);
  const int vecBlocks = N_EMB / 256;

  // ---- image loss: K over S ----
  init_uv<<<vecBlocks, 256, 0, stream>>>(uvec, vvec, N_EMB);
  for (int it = 0; it < 5; ++it) {
    zero_vec<<<vecBlocks, 256, 0, stream>>>(tvec, N_EMB);
    colsum_exp<<<csGrid, 256, 0, stream>>>(Smat, uvec, tvec, +1.f);    // t = K^T u
    vdiv_b<<<vecBlocks, 256, 0, stream>>>(vvec, tvec, N_EMB);          // v = b / t
    rowdot_exp<<<N_EMB / 8, 256, 0, stream>>>(Smat, vvec, uvec, -1.f); // u = Kinv v
  }
  loss_rows<<<N_EMB / 8, 256, 0, stream>>>(Smat, uvec, vvec, out, scale);

  // ---- text loss: matrix is S^T; keep scans of S coalesced by swapping roles ----
  init_uv<<<vecBlocks, 256, 0, stream>>>(uvec, vvec, N_EMB);
  for (int it = 0; it < 5; ++it) {
    rowdot_exp<<<N_EMB / 8, 256, 0, stream>>>(Smat, uvec, tvec, +1.f); // t = K_t^T u = K_img u
    vdiv_b<<<vecBlocks, 256, 0, stream>>>(vvec, tvec, N_EMB);
    zero_vec<<<vecBlocks, 256, 0, stream>>>(uvec, N_EMB);
    colsum_exp<<<csGrid, 256, 0, stream>>>(Smat, vvec, uvec, -1.f);    // u = Kinv_t v
  }
  loss_cols<<<N_EMB / 256, 256, 0, stream>>>(Smat, uvec, vvec, out, scale);
}
